// GraphCastModel_15006615733530
// MI455X (gfx1250) — matrix operand order repaired and checked by reading, not compiled, not run on hardware
//
#include <hip/hip_runtime.h>
#include <hip/hip_bf16.h>

#define NN 65536
#define NE 65536

typedef __attribute__((ext_vector_type(4)))  float        v4f;
typedef __attribute__((ext_vector_type(8)))  float        v8f;
typedef __attribute__((ext_vector_type(16))) __bf16       v16bf;
typedef __attribute__((ext_vector_type(4)))  unsigned int v4u;
typedef __attribute__((ext_vector_type(4)))  unsigned int u32x4;
typedef __attribute__((ext_vector_type(8)))  int          i32x8;
typedef __attribute__((ext_vector_type(4)))  int          i32x4;

#if __has_builtin(__builtin_amdgcn_tensor_load_to_lds) && __has_builtin(__builtin_amdgcn_s_wait_tensorcnt)
#define GEMM_TDM 1
#else
#define GEMM_TDM 0
#endif

union FragU { v16bf v; v4f f[2]; };

__device__ __forceinline__ float wred16(float x) {
    // sum over the 16-lane half (xor 1,2,4,8 stays inside each half of wave32)
    x += __shfl_xor(x, 1);
    x += __shfl_xor(x, 2);
    x += __shfl_xor(x, 4);
    x += __shfl_xor(x, 8);
    return x;
}
__device__ __forceinline__ float silu_f(float x) {
    return x * (1.0f / (1.0f + __expf(-x)));
}
__device__ __forceinline__ unsigned int pack2_bf16(float a, float b) {
    unsigned int ua = __float_as_uint(a), ub = __float_as_uint(b);
    ua = (ua + 0x7FFFu + ((ua >> 16) & 1u)) >> 16;   // RNE
    ub = (ub + 0x7FFFu + ((ub >> 16) & 1u)) >> 16;
    return ua | (ub << 16);
}
__device__ __forceinline__ v4u packv(v4f a, v4f b) {
    v4u o;
    o.x = pack2_bf16(a.x, a.y); o.y = pack2_bf16(a.z, a.w);
    o.z = pack2_bf16(b.x, b.y); o.w = pack2_bf16(b.z, b.w);
    return o;
}

#if GEMM_TDM
// TDM: DMA a 2D bf16 tile [rows][32] (row stride kp elements) from global into
// LDS, inserting 16B of padding after every 64B row -> 80B LDS row stride,
// matching the conflict-free fragment layout.  Issued once per wave (TDM
// ignores EXEC; issue is per-wave).  Tracked by TENSORcnt.
__device__ __forceinline__ void tdm_load_b(const unsigned short* gsrc,
                                           unsigned int ldsaddr, int kp, int rows) {
    unsigned long long ga = (unsigned long long)(size_t)gsrc;
    u32x4 g0;
    g0.x = 1u;                                        // count=1, user descriptor
    g0.y = ldsaddr;                                   // D#.lds_addr
    g0.z = (unsigned int)ga;                          // D#.global_addr[31:0]
    g0.w = (unsigned int)(ga >> 32) | (2u << 30);     // addr[56:32] | type=2
    i32x8 g1;
    g1[0] = (1 << 16)        // data_size = 2 bytes
          | (1 << 20)        // pad_enable
          | (3 << 22)        // pad_interval: 16 DWORDs (64B) stored ...
          | (3 << 25);       // pad_amount:   4 DWORDs (16B) of pad
    g1[1] = (kp & 0xffff) << 16;                      // tensor_dim0[15:0]
    g1[2] = ((kp >> 16) & 0xffff) | ((rows & 0xffff) << 16);  // dim0 hi | dim1 lo
    g1[3] = (32 << 16);                               // tile_dim0 = 32 (dim1 hi = 0)
    g1[4] = rows & 0xffff;                            // tile_dim1 (tile_dim2 = 0)
    g1[5] = kp;                                       // tensor_dim0_stride[31:0]
    g1[6] = 0;
    g1[7] = 0;
    i32x4 z4 = (i32x4)0;
#if __clang_major__ >= 23
    i32x8 z8 = (i32x8)0;
    __builtin_amdgcn_tensor_load_to_lds(g0, g1, z4, z4, z8, 0);
#else
    __builtin_amdgcn_tensor_load_to_lds(g0, g1, z4, z4, 0);
#endif
}
#endif

// ---------------------------------------------------------------------------
// Fused WMMA GEMM. M tile = 128 rows (8 waves x 16 rows), output width NT*16.
// A-tile staging fused with gather/concat/scale/f32->bf16; B tiles DMA'd by
// the Tensor Data Mover (one descriptor per wave slice, double-buffered,
// s_wait_tensorcnt before the barrier).
// ASRC 0: pre-packed bf16 A, row stride KP                 (encoder/identity)
// ASRC 1: A[row] = [ h[gidx[row]] (256) | ea[row] (4) | 0 ]     (sender)
// ASRC 2: A[row] = [ h[row] | agg[row]*inv[row] | ea[row] | 0 ] (receiver)
// ASRC 3: plain f32 A, row stride 256                           (decoder)
// MODE 0: +bias, store f32 (col<nvalid)
// MODE 1: +bias, LN(g1,b1), SiLU, store f32
// MODE 2: +bias, LN, SiLU, atomicAdd out[sidx[row]]  (scatter-aggregate)
// MODE 3: +bias, LN, SiLU -> u; SiLU(u)+hres; LN(g2,b2) -> store
// MODE 4: +bias, + hres(identity), store f32 (col<nvalid)
// ---------------------------------------------------------------------------
template <int NT, int MODE, int KP, int ASRC>
__global__ __launch_bounds__(256)
void gemm_epi(const unsigned short* __restrict__ Abf,
              const float* __restrict__ Af32,
              const float* __restrict__ hsrc,
              const float* __restrict__ aggp,
              const float* __restrict__ invp,
              const float* __restrict__ eat,
              const int* __restrict__ gidx,
              const unsigned short* __restrict__ B,
              const float* __restrict__ bias, int nvalid,
              const float* __restrict__ g1v, const float* __restrict__ b1v,
              const float* __restrict__ g2v, const float* __restrict__ b2v,
              const float* __restrict__ hres, int ldres,
              const int* __restrict__ sidx,
              float* __restrict__ out, int ldo) {
    constexpr int MT = 128, KT = 32, LDA = 40, LDB = 40;  // 80B rows: conflict-free b128
    constexpr int NSTEP = KP / KT;
    constexpr int ACH = MT * 4 / 256;        // A 16B-chunks per thread (=2)
    constexpr int BCH = NT * 16 * 4 / 256;   // B 16B-chunks per thread (fallback path)
    __shared__ __align__(16) unsigned short sA[2][MT * LDA];
    __shared__ __align__(16) unsigned short sB[2][NT * 16 * LDB];

    const int tid = threadIdx.x;
    const int w = tid >> 5, lane = tid & 31, ln = lane & 15, hlf = lane >> 4;
    const int rowBase = blockIdx.x * MT;
    const int rA = tid >> 2;           // row / col index of this thread's chunks
    const int cA = (tid & 3) * 8;      // element offset within the 32-wide K slab

    // per-row constants for the fused A sources (2 rows per thread)
    int   grow[ACH];
    float invr[ACH];
#pragma unroll
    for (int j = 0; j < ACH; j++) {
        grow[j] = 0; invr[j] = 1.0f;
        if constexpr (ASRC == 1) grow[j] = gidx[rowBase + rA + 64 * j];
        if constexpr (ASRC == 2) invr[j] = invp[rowBase + rA + 64 * j];
    }

    // load one 8-element bf16 chunk of A (converting on the fly if needed)
    auto loadAchunk = [&](int j, int k0) -> v4u {
        const int row = rowBase + rA + 64 * j;
        const int k = k0 + cA;
        if constexpr (ASRC == 0) {
            return *(const v4u*)(Abf + (size_t)row * KP + k);
        } else if constexpr (ASRC == 1) {
            v4f a = (v4f)0.0f, b = (v4f)0.0f;
            if (k < 256) {
                const float* p = hsrc + (size_t)grow[j] * 256 + k;
                a = *(const v4f*)p; b = *(const v4f*)(p + 4);
            } else if (k == 256) {
                a = *(const v4f*)(eat + (size_t)row * 4);
            }
            return packv(a, b);
        } else if constexpr (ASRC == 2) {
            v4f a = (v4f)0.0f, b = (v4f)0.0f;
            if (k < 256) {
                const float* p = hsrc + (size_t)row * 256 + k;
                a = *(const v4f*)p; b = *(const v4f*)(p + 4);
            } else if (k < 512) {
                const float* p = aggp + (size_t)row * 256 + (k - 256);
                a = *(const v4f*)p; b = *(const v4f*)(p + 4);
                float s = invr[j];
                a.x *= s; a.y *= s; a.z *= s; a.w *= s;
                b.x *= s; b.y *= s; b.z *= s; b.w *= s;
            } else if (k == 512) {
                a = *(const v4f*)(eat + (size_t)row * 4);
            }
            return packv(a, b);
        } else {  // ASRC 3: plain f32, row stride 256
            const float* p = Af32 + (size_t)row * 256 + k;
            return packv(*(const v4f*)p, *(const v4f*)(p + 4));
        }
    };

    v8f acc[NT];
#pragma unroll
    for (int i = 0; i < NT; i++) acc[i] = (v8f)0.0f;

    // ---- preload tile 0 ----
    v4u ra[ACH];
#pragma unroll
    for (int j = 0; j < ACH; j++) ra[j] = loadAchunk(j, 0);

#if GEMM_TDM
    constexpr int BROWS = NT * 2;          // B columns handled per wave
    const int bcol0 = w * BROWS;
    tdm_load_b(B + (size_t)bcol0 * KP,
               (unsigned int)(size_t)&sB[0][bcol0 * LDB], KP, BROWS);
#else
    v4u rb[BCH];
#pragma unroll
    for (int j = 0; j < BCH; j++)
        rb[j] = *(const v4u*)(B + (size_t)(rA + 64 * j) * KP + cA);
#endif

    int buf = 0;
    for (int s = 0; s < NSTEP; s++) {
        // ---- commit staged A registers to LDS ----
#pragma unroll
        for (int j = 0; j < ACH; j++)
            *(v4u*)&sA[buf][(rA + 64 * j) * LDA + cA] = ra[j];
#if GEMM_TDM
        __builtin_amdgcn_s_wait_tensorcnt(0);   // B tile s landed in sB[buf]
#else
#pragma unroll
        for (int j = 0; j < BCH; j++)
            *(v4u*)&sB[buf][(rA + 64 * j) * LDB + cA] = rb[j];
#endif
        __syncthreads();

        // ---- issue next tile's loads; they retire during the WMMAs ----
        if (s + 1 < NSTEP) {
            const int k0 = (s + 1) * KT;
#pragma unroll
            for (int j = 0; j < ACH; j++) ra[j] = loadAchunk(j, k0);
#if GEMM_TDM
            tdm_load_b(B + (size_t)bcol0 * KP + k0,
                       (unsigned int)(size_t)&sB[buf ^ 1][bcol0 * LDB], KP, BROWS);
#else
#pragma unroll
            for (int j = 0; j < BCH; j++)
                rb[j] = *(const v4u*)(B + (size_t)(rA + 64 * j) * KP + k0 + cA);
#endif
        }

        // A fragment (16x32 bf16): lanes 0-15 K[0..7],[16..23]; lanes 16-31 K[8..15],[24..31]
        FragU a;
        a.f[0] = *(const v4f*)&sA[buf][(w * 16 + ln) * LDA + hlf * 8];
        a.f[1] = *(const v4f*)&sA[buf][(w * 16 + ln) * LDA + hlf * 8 + 16];

        // B fragments: rotate two registers so ds_load(nt+1) overlaps wmma(nt)
        FragU bc, bn;
        bc.f[0] = *(const v4f*)&sB[buf][ln * LDB + hlf * 8];
        bc.f[1] = *(const v4f*)&sB[buf][ln * LDB + hlf * 8 + 16];
        bn = bc;
#pragma unroll
        for (int nt = 0; nt < NT; nt++) {
            if (nt + 1 < NT) {
                bn.f[0] = *(const v4f*)&sB[buf][((nt + 1) * 16 + ln) * LDB + hlf * 8];
                bn.f[1] = *(const v4f*)&sB[buf][((nt + 1) * 16 + ln) * LDB + hlf * 8 + 16];
            }
            acc[nt] = __builtin_amdgcn_wmma_f32_16x16x32_bf16(
                false, a.v, false, bc.v, (short)0, acc[nt], false, false);
            bc = bn;
        }
        buf ^= 1;
    }

    // ---- epilogue: VGPR r holds rows r (lanes 0-15) and r+8 (lanes 16-31), col = nt*16+ln
    float bv[NT];
#pragma unroll
    for (int nt = 0; nt < NT; nt++) {
        int col = nt * 16 + ln;
        bv[nt] = (col < nvalid) ? bias[col] : 0.0f;
    }
#pragma unroll
    for (int nt = 0; nt < NT; nt++)
#pragma unroll
        for (int r = 0; r < 8; r++) acc[nt][r] += bv[nt];

    if constexpr (MODE == 0) {
#pragma unroll
        for (int nt = 0; nt < NT; nt++) {
            int col = nt * 16 + ln;
            if (col < nvalid) {
#pragma unroll
                for (int r = 0; r < 8; r++)
                    out[(size_t)(rowBase + w * 16 + r + 8 * hlf) * ldo + col] = acc[nt][r];
            }
        }
        return;
    }

    if constexpr (MODE == 4) {
#pragma unroll
        for (int nt = 0; nt < NT; nt++) {
            int col = nt * 16 + ln;
            if (col < nvalid) {
#pragma unroll
                for (int r = 0; r < 8; r++) {
                    int row = rowBase + w * 16 + r + 8 * hlf;
                    out[(size_t)row * ldo + col] = acc[nt][r] + hres[(size_t)row * ldres + col];
                }
            }
        }
        return;
    }

    // MODES 1/2/3: LayerNorm over the full 256-wide row, then SiLU
    {
        float gv[NT], bev[NT];
#pragma unroll
        for (int nt = 0; nt < NT; nt++) {
            int col = nt * 16 + ln;
            gv[nt] = g1v[col];
            bev[nt] = b1v[col];
        }
#pragma unroll
        for (int r = 0; r < 8; r++) {
            float s = 0.0f, s2 = 0.0f;
#pragma unroll
            for (int nt = 0; nt < NT; nt++) { float v = acc[nt][r]; s += v; s2 += v * v; }
            s = wred16(s); s2 = wred16(s2);
            const float inv = 1.0f / (float)(NT * 16);
            float mean = s * inv;
            float var = s2 * inv - mean * mean;
            float rstd = rsqrtf(var + 1e-5f);
#pragma unroll
            for (int nt = 0; nt < NT; nt++)
                acc[nt][r] = silu_f((acc[nt][r] - mean) * rstd * gv[nt] + bev[nt]);
        }
    }

    if constexpr (MODE == 1) {
#pragma unroll
        for (int nt = 0; nt < NT; nt++) {
            int col = nt * 16 + ln;
#pragma unroll
            for (int r = 0; r < 8; r++)
                out[(size_t)(rowBase + w * 16 + r + 8 * hlf) * ldo + col] = acc[nt][r];
        }
        return;
    }

    if constexpr (MODE == 2) {
#pragma unroll
        for (int r = 0; r < 8; r++) {
            int e = rowBase + w * 16 + r + 8 * hlf;
            int d = sidx[e];
#pragma unroll
            for (int nt = 0; nt < NT; nt++)
                atomicAdd(out + (size_t)d * ldo + nt * 16 + ln, acc[nt][r]);
        }
        return;
    }

    if constexpr (MODE == 3) {
        // h' = LN2( h + SiLU(u) )
#pragma unroll
        for (int r = 0; r < 8; r++) {
            int row = rowBase + w * 16 + r + 8 * hlf;
#pragma unroll
            for (int nt = 0; nt < NT; nt++) {
                int col = nt * 16 + ln;
                acc[nt][r] = silu_f(acc[nt][r]) + hres[(size_t)row * ldres + col];
            }
        }
        float gv[NT], bev[NT];
#pragma unroll
        for (int nt = 0; nt < NT; nt++) {
            int col = nt * 16 + ln;
            gv[nt] = g2v[col];
            bev[nt] = b2v[col];
        }
#pragma unroll
        for (int r = 0; r < 8; r++) {
            float s = 0.0f, s2 = 0.0f;
#pragma unroll
            for (int nt = 0; nt < NT; nt++) { float v = acc[nt][r]; s += v; s2 += v * v; }
            s = wred16(s); s2 = wred16(s2);
            const float inv = 1.0f / (float)(NT * 16);
            float mean = s * inv;
            float var = s2 * inv - mean * mean;
            float rstd = rsqrtf(var + 1e-5f);
            int row = rowBase + w * 16 + r + 8 * hlf;
#pragma unroll
            for (int nt = 0; nt < NT; nt++)
                out[(size_t)row * ldo + nt * 16 + ln] =
                    (acc[nt][r] - mean) * rstd * gv[nt] + bev[nt];
        }
    }
}

// ---------------------------------------------------------------------------
// Prep / utility kernels
// ---------------------------------------------------------------------------
__global__ void zero_f32(float* p, int n4) {
    int i = blockIdx.x * 256 + threadIdx.x;
    if (i < n4) ((v4f*)p)[i] = (v4f)0.0f;
}
__global__ void deg_count(const int* __restrict__ dst, float* __restrict__ deg) {
    int e = blockIdx.x * 256 + threadIdx.x;
    if (e < NE) atomicAdd(&deg[dst[e]], 1.0f);
}
__global__ void inv_deg_k(float* d) {
    int i = blockIdx.x * 256 + threadIdx.x;
    if (i < NN) d[i] = 1.0f / fmaxf(d[i], 1.0f);
}
// W f32 [K][Ncols] row-major  ->  bf16 [Npad][Kpad] (K contiguous per col, zero pad)
__global__ void pack_weight(const float* __restrict__ W, unsigned short* __restrict__ out,
                            int K, int Ncols, int Kpad, int Npad) {
    int t = blockIdx.x * 256 + threadIdx.x;
    int total = Npad * Kpad;
    if (t >= total) return;
    int c = t / Kpad, k = t % Kpad;
    float v = (k < K && c < Ncols) ? W[(size_t)k * Ncols + c] : 0.0f;
    unsigned int u = __float_as_uint(v);
    u = (u + 0x7FFFu + ((u >> 16) & 1u)) >> 16;
    out[(size_t)c * Kpad + k] = (unsigned short)u;
}
// f32 [rows][srcC] -> bf16 [rows][dstC] zero padded (node features -> 96)
__global__ void pack_pad(const float* __restrict__ src, unsigned short* __restrict__ dst,
                         int rows, int srcC, int dstC) {
    int t = blockIdx.x * 256 + threadIdx.x;
    int total = rows * dstC;
    if (t >= total) return;
    int r = t / dstC, c = t % dstC;
    float v = (c < srcC) ? src[(size_t)r * srcC + c] : 0.0f;
    unsigned int u = __float_as_uint(v);
    u = (u + 0x7FFFu + ((u >> 16) & 1u)) >> 16;
    dst[t] = (unsigned short)u;
}

// ---------------------------------------------------------------------------
extern "C" void kernel_launch(void* const* d_in, const int* in_sizes, int n_in,
                              void* d_out, int out_size, void* d_ws, size_t ws_size,
                              hipStream_t stream) {
    const float* x        = (const float*)d_in[0];
    const int*   eidx     = (const int*)d_in[1];
    const float* ea       = (const float*)d_in[2];
    const float* enc_W    = (const float*)d_in[3];
    const float* enc_b    = (const float*)d_in[4];
    const float* enc_g    = (const float*)d_in[5];
    const float* enc_beta = (const float*)d_in[6];
    const float* Ws       = (const float*)d_in[7];
    const float* bs       = (const float*)d_in[8];
    const float* gs       = (const float*)d_in[9];
    const float* betas    = (const float*)d_in[10];
    const float* Wr       = (const float*)d_in[11];
    const float* br       = (const float*)d_in[12];
    const float* gr       = (const float*)d_in[13];
    const float* betar    = (const float*)d_in[14];
    const float* g1       = (const float*)d_in[15];
    const float* beta1    = (const float*)d_in[16];
    const float* dW1      = (const float*)d_in[17];
    const float* db1      = (const float*)d_in[18];
    const float* dg       = (const float*)d_in[19];
    const float* dbeta    = (const float*)d_in[20];
    const float* dW2      = (const float*)d_in[21];
    const float* db2      = (const float*)d_in[22];
    const float* rW       = (const float*)d_in[23];
    const float* rb       = (const float*)d_in[24];
    const int* srcI = eidx;
    const int* dstI = eidx + NE;

    char* wsb = (char*)d_ws;
    size_t cur = 0;
    auto alloc = [&](size_t b) { size_t p = cur; cur = (cur + b + 255) & ~(size_t)255; return p; };
    unsigned short* pWenc = (unsigned short*)(wsb + alloc((size_t)96 * 256 * 2));
    unsigned short* pWres = (unsigned short*)(wsb + alloc((size_t)96 * 128 * 2));
    unsigned short* pWsnd = (unsigned short*)(wsb + alloc((size_t)16 * 288 * 256 * 2));
    unsigned short* pWrcv = (unsigned short*)(wsb + alloc((size_t)16 * 544 * 256 * 2));
    unsigned short* pWd1  = (unsigned short*)(wsb + alloc((size_t)256 * 256 * 2));
    unsigned short* pWd2  = (unsigned short*)(wsb + alloc((size_t)256 * 128 * 2));
    float* pInv   = (float*)(wsb + alloc((size_t)NN * 4));
    float* pIdent = (float*)(wsb + alloc((size_t)NN * 128 * 4));
    float* ph0    = (float*)(wsb + alloc((size_t)NN * 256 * 4));
    float* ph1    = (float*)(wsb + alloc((size_t)NN * 256 * 4));
    float* pAgg   = (float*)(wsb + alloc((size_t)NN * 256 * 4));
    unsigned short* pXbf = (unsigned short*)(wsb + alloc((size_t)NN * 96 * 2));

    auto cdiv = [](long a, long b) { return (int)((a + b - 1) / b); };
    const unsigned short* nbf = nullptr;
    const float* nf = nullptr;
    const int* ni = nullptr;

    // ---- pack weights (bf16, [N][Kpad], zero padded) ----
    pack_weight<<<cdiv(256 * 96, 256), 256, 0, stream>>>(enc_W, pWenc, 70, 256, 96, 256);
    pack_weight<<<cdiv(128 * 96, 256), 256, 0, stream>>>(rW, pWres, 70, 70, 96, 128);
    for (int l = 0; l < 16; l++) {
        pack_weight<<<cdiv(256 * 288, 256), 256, 0, stream>>>(
            Ws + (size_t)l * 260 * 256, pWsnd + (size_t)l * 288 * 256, 260, 256, 288, 256);
        pack_weight<<<cdiv(256 * 544, 256), 256, 0, stream>>>(
            Wr + (size_t)l * 516 * 256, pWrcv + (size_t)l * 544 * 256, 516, 256, 544, 256);
    }
    pack_weight<<<cdiv(256 * 256, 256), 256, 0, stream>>>(dW1, pWd1, 256, 256, 256, 256);
    pack_weight<<<cdiv(128 * 256, 256), 256, 0, stream>>>(dW2, pWd2, 256, 70, 256, 128);

    // ---- inverse degree ----
    zero_f32<<<cdiv(NN / 4, 256), 256, 0, stream>>>(pInv, NN / 4);
    deg_count<<<cdiv(NE, 256), 256, 0, stream>>>(dstI, pInv);
    inv_deg_k<<<cdiv(NN, 256), 256, 0, stream>>>(pInv);

    // ---- encoder + residual skip (x packed once to bf16 [N,96]) ----
    pack_pad<<<cdiv((long)NN * 96, 256), 256, 0, stream>>>(x, pXbf, NN, 70, 96);
    gemm_epi<8, 0, 96, 0><<<NN / 128, 256, 0, stream>>>(
        pXbf, nf, nf, nf, nf, nf, ni, pWres, rb, 70,
        nf, nf, nf, nf, nf, 0, ni, pIdent, 128);
    gemm_epi<16, 1, 96, 0><<<NN / 128, 256, 0, stream>>>(
        pXbf, nf, nf, nf, nf, nf, ni, pWenc, enc_b, 256,
        enc_g, enc_beta, nf, nf, nf, 0, ni, ph0, 256);

    // ---- 16 message-passing layers (fully fused: no A staging in memory) ----
    float* hc = ph0;
    float* hn = ph1;
    for (int l = 0; l < 16; l++) {
        zero_f32<<<cdiv((long)NN * 256 / 4, 256), 256, 0, stream>>>(pAgg, NN * 256 / 4);
        // sender: gather h[src] + edge_attr -> MLP -> LN -> SiLU -> atomic scatter
        gemm_epi<16, 2, 288, 1><<<NE / 128, 256, 0, stream>>>(
            nbf, nf, hc, nf, nf, ea, srcI,
            pWsnd + (size_t)l * 288 * 256, bs + (size_t)l * 256, 256,
            gs + (size_t)l * 256, betas + (size_t)l * 256, nf, nf,
            nf, 0, dstI, pAgg, 256);
        // receiver: [h | agg/deg | ea] -> MLP -> LN -> SiLU -> +h -> LN
        gemm_epi<16, 3, 544, 2><<<NN / 128, 256, 0, stream>>>(
            nbf, nf, hc, pAgg, pInv, ea, ni,
            pWrcv + (size_t)l * 544 * 256, br + (size_t)l * 256, 256,
            gr + (size_t)l * 256, betar + (size_t)l * 256,
            g1 + (size_t)l * 256, beta1 + (size_t)l * 256,
            hc, 256, ni, hn, 256);
        float* t = hc; hc = hn; hn = t;
    }

    // ---- decoder + skip (reads f32 h directly) ----
    gemm_epi<16, 1, 256, 3><<<NN / 128, 256, 0, stream>>>(
        nbf, hc, nf, nf, nf, nf, ni, pWd1, db1, 256,
        dg, dbeta, nf, nf, nf, 0, ni, pAgg, 256);
    gemm_epi<8, 4, 256, 3><<<NN / 128, 256, 0, stream>>>(
        nbf, pAgg, nf, nf, nf, nf, ni, pWd2, db2, 70,
        nf, nf, nf, nf, pIdent, 128, ni, (float*)d_out, 70);

    (void)in_sizes; (void)n_in; (void)out_size; (void)ws_size;
}
